// AddBeta_64080912056949
// MI455X (gfx1250) — matrix operand order repaired and checked by reading, not compiled, not run on hardware
//
#include <hip/hip_runtime.h>

// ---------------------------------------------------------------------------
// Sizes (from reference)
// ---------------------------------------------------------------------------
#define BS        512
#define D_IN      256
#define XROW      257      // x row length (OUTPUT_LEN + 1)
#define H1N       100
#define H2N       100
#define LAT       512
#define ENCN      1024     // 2*LAT
#define G3        1536     // 3*LAT
#define TOUT      256
#define BN_EPS    1e-5f

typedef __attribute__((ext_vector_type(16))) _Float16 v16h;
typedef __attribute__((ext_vector_type(8)))  float    v8f;
typedef __attribute__((ext_vector_type(2)))  _Float16 h2;

// ---------------------------------------------------------------------------
// Zero the loss accumulator (must re-run every launch: graph replay safe)
// ---------------------------------------------------------------------------
__global__ void k_zero(float* acc) { acc[0] = 0.0f; }

// ---------------------------------------------------------------------------
// Generic small dense: out[b,f] = (relu?)(in[b,:] @ W + bias)
// grid = (BS, ceil(N/128)), block = 128
// ---------------------------------------------------------------------------
__global__ __launch_bounds__(128)
void k_dense(const float* __restrict__ in, int rstride, int K,
             const float* __restrict__ W, int N,
             const float* __restrict__ bias,
             float* __restrict__ out, int do_relu) {
  const int b = blockIdx.x;
  const int f = blockIdx.y * blockDim.x + threadIdx.x;
  if (f >= N) return;
  const float* row = in + b * rstride;
  float acc = bias[f];
  for (int k = 0; k < K; ++k)
    acc = fmaf(row[k], W[k * N + f], acc);
  if (do_relu) acc = fmaxf(acc, 0.0f);
  out[b * N + f] = acc;
}

// ---------------------------------------------------------------------------
// Batchnorm over batch axis, in place. grid = (N features), block = 256.
// Each block owns one feature column (512 rows, 2 per thread).
// ---------------------------------------------------------------------------
__global__ __launch_bounds__(256)
void k_bn(float* __restrict__ h, int N,
          const float* __restrict__ g, const float* __restrict__ bb) {
  const int f = blockIdx.x;
  const int tid = threadIdx.x;
  const float v0 = h[tid * N + f];
  const float v1 = h[(tid + 256) * N + f];
  float s = v0 + v1, ss = v0 * v0 + v1 * v1;
  for (int off = 16; off; off >>= 1) {
    s  += __shfl_down(s,  off, 32);
    ss += __shfl_down(ss, off, 32);
  }
  __shared__ float rs[8], rss[8];
  __shared__ float s_mean, s_scale;
  const int lane = tid & 31, wid = tid >> 5;
  if (lane == 0) { rs[wid] = s; rss[wid] = ss; }
  __syncthreads();
  if (tid == 0) {
    float S = 0.f, SS = 0.f;
    for (int w = 0; w < 8; ++w) { S += rs[w]; SS += rss[w]; }
    const float mean = S * (1.0f / 512.0f);
    const float var  = SS * (1.0f / 512.0f) - mean * mean;  // ddof=0, as jnp.var
    s_mean  = mean;
    s_scale = rsqrtf(var + BN_EPS) * g[f];
  }
  __syncthreads();
  const float beta = bb[f];
  h[tid * N + f]         = (v0 - s_mean) * s_scale + beta;
  h[(tid + 256) * N + f] = (v1 - s_mean) * s_scale + beta;
}

// ---------------------------------------------------------------------------
// Reparameterization + KL partial sums.
// z (f32) -> d_out slot, z (f16) -> scratch for WMMA. grid=BS, block=256.
// ---------------------------------------------------------------------------
__global__ __launch_bounds__(256)
void k_latent(const float* __restrict__ enc, const float* __restrict__ eps,
              float* __restrict__ z_out, _Float16* __restrict__ zh,
              float* __restrict__ acc) {
  const int b = blockIdx.x, tid = threadIdx.x;
  float part = 0.0f;
  for (int h = tid; h < LAT; h += 256) {
    const float mu = enc[b * ENCN + h];
    const float lv = enc[b * ENCN + LAT + h];
    const float ex = __expf(lv);
    const float z  = fmaf(eps[b * LAT + h], __expf(0.5f * lv), mu);
    z_out[b * LAT + h] = z;
    zh[b * LAT + h]    = (_Float16)z;
    part += 1.0f + lv - mu * mu - ex;
  }
  for (int off = 16; off; off >>= 1) part += __shfl_down(part, off, 32);
  __shared__ float red[8];
  const int lane = tid & 31, wid = tid >> 5;
  if (lane == 0) red[wid] = part;
  __syncthreads();
  if (tid == 0) {
    float s = 0.f;
    for (int w = 0; w < 8; ++w) s += red[w];
    atomicAdd(acc, s);
  }
}

__global__ void k_loss(const float* acc, float* loss) {
  loss[0] = -0.5f * acc[0] * (1.0f / (float)BS);
}

// ---------------------------------------------------------------------------
// f32 -> f16 conversion for W_hh (every launch; no cached state)
// ---------------------------------------------------------------------------
__global__ __launch_bounds__(256)
void k_half(const float* __restrict__ src, _Float16* __restrict__ dst, int n) {
  const int i = blockIdx.x * blockDim.x + threadIdx.x;
  if (i < n) dst[i] = (_Float16)src[i];
}

// ---------------------------------------------------------------------------
// gh = z @ W_hh^T + b_hh via v_wmma_f32_16x16x32_f16, register-blocked:
// each wave computes a 16(M) x 64(N) strip with 4 accumulators. Per 32-wide
// K-chunk: gather ALL fragments (A + 4xB, 10 b128 loads) into registers
// first, then issue the 4 WMMAs, so waits are staged and the scheduler can
// hoist the next chunk's loads above the current WMMAs (unroll 2).
// Prefetch one full 128B line (2 K-chunks) ahead per stream.
// grid = (G3/64/4 = 6, BS/16 = 32), block = 128 (4 waves).
// Lane layouts per CDNA5 ISA 7.12.2 (16-bit A 16x32, 16-bit B 32x16, f32 C/D).
// ---------------------------------------------------------------------------
__global__ __launch_bounds__(128)
void k_gh_wmma(const _Float16* __restrict__ zh,   // 512 x 512
               const _Float16* __restrict__ wh,   // 1536 x 512 (row-major)
               const float* __restrict__ bhh,     // 1536
               float* __restrict__ gh) {          // 512 x 1536
  const int lane = threadIdx.x & 31;
  const int wv   = threadIdx.x >> 5;
  const int m0 = blockIdx.y * 16;
  const int n0 = (blockIdx.x * 4 + wv) * 64;     // 4 N-tiles of 16
  const int hi  = lane >> 4;
  const int low = lane & 15;

  const _Float16* arow = zh + (m0 + low) * LAT;  // A row for this lane
  const _Float16* brow = wh + (n0 + low) * LAT;  // B col (tile j: +j*16*LAT)

  v8f c0 = {}, c1 = {}, c2 = {}, c3 = {};
#pragma unroll 2
  for (int k0 = 0; k0 < LAT; k0 += 32) {
    // Prefetch a full cacheline (2 K-chunks) ahead of each stream
    // (speculative prefetch; silently dropped past buffer end).
    __builtin_prefetch(arow + k0 + 64, 0, 3);
    __builtin_prefetch(brow + k0 + 64, 0, 3);
    __builtin_prefetch(brow + 16 * LAT + k0 + 64, 0, 3);
    __builtin_prefetch(brow + 32 * LAT + k0 + 64, 0, 3);
    __builtin_prefetch(brow + 48 * LAT + k0 + 64, 0, 3);

    // ---- Gather all fragments first ----
    // A 16x32 f16: VGPR vg holds K = {base + 2*(vg&3), +1}; lanes 16-31 have
    // base +8; VGPRs 4-7 cover K 16..31.
    v16h a;
#pragma unroll
    for (int vg = 0; vg < 8; ++vg) {
      const int ka = ((vg & 4) ? 16 : 0) + hi * 8 + (vg & 3) * 2;
      const h2 pa = *(const h2*)(arow + k0 + ka);
      a[2 * vg]     = pa[0];
      a[2 * vg + 1] = pa[1];
    }
    // B 32x16 f16: same K layout as A (col = lane & 15).
    v16h b0, b1, b2, b3;
#pragma unroll
    for (int vg = 0; vg < 8; ++vg) {
      const int kb = ((vg & 4) ? 16 : 0) + hi * 8 + (vg & 3) * 2;
      const h2 p0 = *(const h2*)(brow + 0 * 16 * LAT + k0 + kb);
      const h2 p1 = *(const h2*)(brow + 1 * 16 * LAT + k0 + kb);
      const h2 p2 = *(const h2*)(brow + 2 * 16 * LAT + k0 + kb);
      const h2 p3 = *(const h2*)(brow + 3 * 16 * LAT + k0 + kb);
      b0[2 * vg] = p0[0]; b0[2 * vg + 1] = p0[1];
      b1[2 * vg] = p1[0]; b1[2 * vg + 1] = p1[1];
      b2[2 * vg] = p2[0]; b2[2 * vg + 1] = p2[1];
      b3[2 * vg] = p3[0]; b3[2 * vg + 1] = p3[1];
    }
    // ---- Then compute ----
    c0 = __builtin_amdgcn_wmma_f32_16x16x32_f16(false, a, false, b0, (short)0, c0, false, false);
    c1 = __builtin_amdgcn_wmma_f32_16x16x32_f16(false, a, false, b1, (short)0, c1, false, false);
    c2 = __builtin_amdgcn_wmma_f32_16x16x32_f16(false, a, false, b2, (short)0, c2, false, false);
    c3 = __builtin_amdgcn_wmma_f32_16x16x32_f16(false, a, false, b3, (short)0, c3, false, false);
  }
  // C/D layout: VGPR r -> M = hi*8 + r, N = lane & 15
#pragma unroll
  for (int j = 0; j < 4; ++j) {
    const v8f c = (j == 0) ? c0 : (j == 1) ? c1 : (j == 2) ? c2 : c3;
    const int n = n0 + j * 16 + low;
    const float bias = bhh[n];
#pragma unroll
    for (int r = 0; r < 8; ++r)
      gh[(m0 + hi * 8 + r) * G3 + n] = c[r] + bias;
  }
}

// ---------------------------------------------------------------------------
// Fused GRU-cell + ReLU + fc contraction + weights epilogue.
// One block (256 threads) per batch row; gh/z/w_ih/b_ih/fc_w staged in LDS
// (20 KB of the WGP's 320 KB). hnew is never materialized.
// ---------------------------------------------------------------------------
__global__ __launch_bounds__(256)
void k_gru(const float* __restrict__ x,       // 512 x 1 x 257
           const float* __restrict__ zf,      // 512 x 512 (f32, from d_out)
           const float* __restrict__ gh,      // 512 x 1536
           const float* __restrict__ w_ih,    // 1536
           const float* __restrict__ b_ih,    // 1536
           const float* __restrict__ fc_w,    // 512 x 2
           const float* __restrict__ fc_b,    // 2
           const float* __restrict__ kbin,    // 256
           const float* __restrict__ kact,    // 256
           float* __restrict__ predict) {     // 512 x 1 x 2 x 256
  __shared__ float s_wr[LAT], s_wz[LAT], s_wn[LAT];
  __shared__ float s_cr[LAT], s_cz[LAT], s_bn[LAT], s_gn[LAT];
  __shared__ float s_z[LAT], s_f0[LAT], s_f1[LAT];
  __shared__ float red0[8], red1[8];

  const int b = blockIdx.x, tid = threadIdx.x;
  for (int h = tid; h < LAT; h += 256) {
    s_wr[h] = w_ih[h];
    s_wz[h] = w_ih[LAT + h];
    s_wn[h] = w_ih[2 * LAT + h];
    s_cr[h] = gh[b * G3 + h]           + b_ih[h];            // r pre-act const
    s_cz[h] = gh[b * G3 + LAT + h]     + b_ih[LAT + h];      // z pre-act const
    s_gn[h] = gh[b * G3 + 2 * LAT + h];                      // gh_n (gated by r)
    s_bn[h] = b_ih[2 * LAT + h];
    s_z[h]  = zf[b * LAT + h];
    s_f0[h] = fc_w[2 * h];
    s_f1[h] = fc_w[2 * h + 1];
  }
  __syncthreads();

  const int lane = tid & 31, wid = tid >> 5;
  for (int t = 0; t < TOUT; ++t) {
    const float xv = x[b * XROW + t];
    float p0 = 0.0f, p1 = 0.0f;
#pragma unroll
    for (int i = 0; i < 2; ++i) {
      const int h = tid + 256 * i;
      const float r  = 1.0f / (1.0f + __expf(-fmaf(xv, s_wr[h], s_cr[h])));
      const float zt = 1.0f / (1.0f + __expf(-fmaf(xv, s_wz[h], s_cz[h])));
      const float n  = tanhf(fmaf(xv, s_wn[h], s_bn[h]) + r * s_gn[h]);
      const float hn = (1.0f - zt) * n + zt * s_z[h];
      const float hr = fmaxf(hn, 0.0f);
      p0 = fmaf(hr, s_f0[h], p0);
      p1 = fmaf(hr, s_f1[h], p1);
    }
    for (int off = 16; off; off >>= 1) {
      p0 += __shfl_down(p0, off, 32);
      p1 += __shfl_down(p1, off, 32);
    }
    if (lane == 0) { red0[wid] = p0; red1[wid] = p1; }
    __syncthreads();
    if (tid == 0) {
      float t0 = fc_b[0], t1 = fc_b[1];
      for (int w = 0; w < 8; ++w) { t0 += red0[w]; t1 += red1[w]; }
      // offset(): base * gaussian window; l = T_CROSS/5 = 4, 2*l*l = 32
      const float base = (t & 1) ? 0.2f : 0.5f;
      const float j = (float)(t >> 2);
      const float win = (t & 2) ? __expf(-(j * j) * (1.0f / 32.0f))
                                : __expf(-((j - 19.0f) * (j - 19.0f)) * (1.0f / 32.0f));
      const float off_t = base * win;
      const float eff = (t >= 20) ? -1.0f : 1.0f;
      const float kb = kbin[t], ka = kact[t];
      const float spb = (kb > 20.0f) ? kb : log1pf(__expf(kb));
      const float spa = (ka > 20.0f) ? ka : log1pf(__expf(ka));
      predict[b * 512 + t]        = t0 + eff * (spb + off_t);  // channel 0
      predict[b * 512 + 256 + t]  = t1 + eff * (spa + off_t);  // channel 1
    }
    __syncthreads();
  }
}

// ---------------------------------------------------------------------------
// Launch
// ---------------------------------------------------------------------------
extern "C" void kernel_launch(void* const* d_in, const int* in_sizes, int n_in,
                              void* d_out, int out_size, void* d_ws, size_t ws_size,
                              hipStream_t stream) {
  const float* x     = (const float*)d_in[0];
  const float* eps   = (const float*)d_in[1];
  const float* w1    = (const float*)d_in[2];
  const float* b1    = (const float*)d_in[3];
  const float* bn1g  = (const float*)d_in[4];
  const float* bn1b  = (const float*)d_in[5];
  const float* w2    = (const float*)d_in[6];
  const float* b2    = (const float*)d_in[7];
  const float* bn2g  = (const float*)d_in[8];
  const float* bn2b  = (const float*)d_in[9];
  const float* w3    = (const float*)d_in[10];
  const float* b3    = (const float*)d_in[11];
  const float* w_ih  = (const float*)d_in[12];
  const float* w_hh  = (const float*)d_in[13];
  const float* b_ih  = (const float*)d_in[14];
  const float* b_hh  = (const float*)d_in[15];
  const float* fcw   = (const float*)d_in[16];
  const float* fcb   = (const float*)d_in[17];
  const float* kbin  = (const float*)d_in[18];
  const float* kact  = (const float*)d_in[19];

  float* out     = (float*)d_out;
  float* predict = out;                       // 512*2*256 = 262144
  float* loss    = out + 262144;              // 1
  float* zf      = out + 262145;              // 512*512 = 262144

  // Workspace layout (~7.5 MB)
  float* H1   = (float*)d_ws;                 // 512*100
  float* H2   = H1 + BS * H1N;                // 512*100
  float* enc  = H2 + BS * H2N;                // 512*1024
  float* gh   = enc + BS * ENCN;              // 512*1536
  float* acc  = gh + BS * G3;                 // 1 (+3 pad)
  _Float16* zh   = (_Float16*)(acc + 4);      // 512*512 halves
  _Float16* wh16 = zh + BS * LAT;             // 1536*512 halves

  k_zero<<<1, 1, 0, stream>>>(acc);

  // Encoder
  k_dense<<<dim3(BS, 1), 128, 0, stream>>>(x,  XROW, D_IN, w1, H1N, b1, H1, 1);
  k_bn   <<<H1N, 256, 0, stream>>>(H1, H1N, bn1g, bn1b);
  k_dense<<<dim3(BS, 1), 128, 0, stream>>>(H1, H1N, H1N, w2, H2N, b2, H2, 1);
  k_bn   <<<H2N, 256, 0, stream>>>(H2, H2N, bn2g, bn2b);
  k_dense<<<dim3(BS, ENCN / 128), 128, 0, stream>>>(H2, H2N, H2N, w3, ENCN, b3, enc, 0);

  // Reparameterize + KL
  k_latent<<<BS, 256, 0, stream>>>(enc, eps, zf, zh, acc);
  k_loss  <<<1, 1, 0, stream>>>(acc, loss);

  // gh = z @ W_hh^T + b_hh  (WMMA f16 -> f32, register-blocked 16x64)
  k_half<<<(G3 * LAT + 255) / 256, 256, 0, stream>>>(w_hh, wh16, G3 * LAT);
  k_gh_wmma<<<dim3(G3 / 64 / 4, BS / 16), 128, 0, stream>>>(zh, wh16, b_hh, gh);

  // Fused GRU + output head
  k_gru<<<BS, 256, 0, stream>>>(x, zf, gh, w_ih, b_ih, fcw, fcb, kbin, kact, predict);
}
